// MS_WSA_6322191860268
// MI455X (gfx1250) — matrix operand order repaired and checked by reading, not compiled, not run on hardware
//
#include <hip/hip_runtime.h>
#include <math.h>
#include <cstdint>

// ---------------------------------------------------------------------------
// MS_WSA transformer block for MI455X (gfx1250, wave32, WMMA).
// - All GEMMs: V_WMMA_F32_16X16X32_BF16 (bf16 in, f32 accumulate)
// - Global->LDS staging: global_load_async_to_lds_b128 + s_wait_asynccnt,
//   double-buffered tiles (memory overlapped with WMMA).
// Requires ws_size >= ~405 MB.
// ---------------------------------------------------------------------------

typedef __attribute__((ext_vector_type(16))) __bf16 bf16x16;
typedef __attribute__((ext_vector_type(8)))  __bf16 bf16x8;
typedef __attribute__((ext_vector_type(8)))  float  f32x8;

#define NWIN   2048
#define LTOK   64
#define CDIM   256
#define HEADS  8
#define DHEAD  32
#define ATOK   48
#define NGROUP 8
#define SCALEF 0.17677669529663687f  // 32^-0.5

// ---------------- async global->LDS helpers (CDNA5) ------------------------
__device__ static inline void async_ld_b128(void* lds_dst, const void* gsrc) {
  unsigned l = (unsigned)(uintptr_t)lds_dst;   // low 32 bits = LDS offset
  asm volatile("global_load_async_to_lds_b128 %0, %1, off"
               :: "v"(l), "v"(gsrc) : "memory");
}
__device__ static inline void wait_async_0() {
  asm volatile("s_wait_asynccnt 0x0" ::: "memory");
}
__device__ static inline void wait_async_4() {   // allow 4 newest in flight
  asm volatile("s_wait_asynccnt 0x4" ::: "memory");
}
__device__ static inline void wait_async_5() {   // allow 5 newest in flight
  asm volatile("s_wait_asynccnt 0x5" ::: "memory");
}

// ---------------- WMMA fragment loaders (k = 16*(lane/16) + j, same order in A and B) ---------
// A fragment: 16x32 bf16 tile from LDS, row-major, stride ld (elements).
__device__ static inline bf16x16 load_a_frag(const __bf16* base, int ld) {
  int lane = threadIdx.x & 31;
  int r    = lane & 15;
  int kh   = (lane >> 4) ? 16 : 0;
  const __bf16* p = base + r * ld;
  bf16x16 a;
#pragma unroll
  for (int j = 0; j < 16; ++j) a[j] = p[kh + j];
  return a;
}
// B fragment: 32x16 (KxN) from LDS stored transposed Bt[n][k], stride ld.
__device__ static inline bf16x16 load_b_frag_t(const __bf16* baseT, int ld) {
  int lane = threadIdx.x & 31;
  int n    = lane & 15;
  int k0   = (lane >> 4) ? 16 : 0;
  const __bf16* p = baseT + n * ld + k0;
  bf16x16 b;
#pragma unroll
  for (int j = 0; j < 16; ++j) b[j] = p[j];
  return b;
}
// B fragment from row-major KxN LDS tile: element (k,n) at base[k*ld+n].
__device__ static inline bf16x16 load_b_frag_rm(const __bf16* base, int ld) {
  int lane = threadIdx.x & 31;
  int n    = lane & 15;
  int k0   = (lane >> 4) ? 16 : 0;
  bf16x16 b;
#pragma unroll
  for (int j = 0; j < 16; ++j) b[j] = base[(k0 + j) * ld + n];
  return b;
}
__device__ static inline f32x8 wmma_bf16(bf16x16 a, bf16x16 b, f32x8 c) {
  return __builtin_amdgcn_wmma_f32_16x16x32_bf16(false, a, false, b,
                                                 (short)0, c, false, false);
}

// ---------------- K0: f32 -> bf16 weight convert + transpose ---------------
// src: row-major [Kd][Nd] f32;  dst: [Nd][Kd] bf16 (so GEMM B-tiles are
// contiguous along k for b128 staging).
__global__ __launch_bounds__(256) void cvt_transpose_kernel(
    const float* __restrict__ src, __bf16* __restrict__ dst, int Kd, int Nd) {
  int i = blockIdx.x * 256 + threadIdx.x;
  if (i < Kd * Nd) {
    int n = i / Kd, k = i - n * Kd;
    dst[i] = (__bf16)src[(size_t)k * Nd + n];
  }
}

// ---------------- K1: LN1(+LN2) producer -----------------------------------
__device__ static inline float block_reduce_sum(float v, float* red) {
  int tid = threadIdx.x;
  red[tid] = v;
  __syncthreads();
#pragma unroll
  for (int s = 128; s > 0; s >>= 1) {
    if (tid < s) red[tid] += red[tid + s];
    __syncthreads();
  }
  float r = red[0];
  __syncthreads();
  return r;
}

__global__ __launch_bounds__(256) void ln_gather_kernel(
    const float* __restrict__ x, const int* __restrict__ iw,
    const float* __restrict__ g1, const float* __restrict__ b1,
    const float* __restrict__ g2, const float* __restrict__ b2,
    float* __restrict__ X_out, float* __restrict__ shortcut,
    __bf16* __restrict__ xt) {
  __shared__ float red[256];
  int id = blockIdx.x;
  int m = id >> 6, t = id & 63;
  int c = threadIdx.x;
  int w = iw[m];
  float v = x[(((size_t)w << 6) + t) * CDIM + c];

  float mu  = block_reduce_sum(v, red) * (1.0f / 256.0f);
  float d   = v - mu;
  float var = block_reduce_sum(d * d, red) * (1.0f / 256.0f);
  float y   = d * rsqrtf(var + 1e-5f) * g1[c] + b1[c];

  if (t >= ATOK) {
    X_out[(((size_t)w << 6) + t) * CDIM + c] = y;   // final value
  } else {
    float mu2  = block_reduce_sum(y, red) * (1.0f / 256.0f);
    float d2   = y - mu2;
    float var2 = block_reduce_sum(d2 * d2, red) * (1.0f / 256.0f);
    y = d2 * rsqrtf(var2 + 1e-5f) * g2[c] + b2[c];
    shortcut[((size_t)m * ATOK + t) * CDIM + c] = y;
  }
  int q = (t - (m & 63) + 64) & 63;                  // (t - m) mod 64
  xt[(((size_t)m << 6) + q) * CDIM + c] = (__bf16)y;
}

// ---------------- K2: fused QKV-GEMM + masked-softmax attention ------------
// grid (M, H), 128 threads (4 wave32).  wqkvT is bf16 [768][256].
__global__ __launch_bounds__(128) void attn_kernel(
    const __bf16* __restrict__ xt, const __bf16* __restrict__ wqkvT,
    const float* __restrict__ qkvb, __bf16* __restrict__ o_asy) {
  const int m = blockIdx.x, h = blockIdx.y;
  const int tid = threadIdx.x, lane = tid & 31, wv = tid >> 5;

  __shared__ __bf16 sX[2][64][40];     // x tile, double buffered
  __shared__ __bf16 sWt[2][96][40];    // Wt[n][k] tile, double buffered
  __shared__ __bf16 sQKV[64][104];
  __shared__ float  sS[64][72];
  __shared__ __bf16 sP[64][72];

  auto stage = [&](int buf, int k0) {
#pragma unroll
    for (int i = 0; i < 2; ++i) {      // 64 rows x 4 chunks of 8 bf16
      int ch = i * 128 + tid, r = ch >> 2, cc = (ch & 3) * 8;
      async_ld_b128(&sX[buf][r][cc],
                    &xt[(((size_t)m << 6) + r) * CDIM + k0 + cc]);
    }
#pragma unroll
    for (int i = 0; i < 3; ++i) {      // 96 n-rows x 4 chunks of 8 bf16
      int ch = i * 128 + tid, n = ch >> 2, kk = (ch & 3) * 8;
      async_ld_b128(&sWt[buf][n][kk],
                    &wqkvT[(size_t)(h * 96 + n) * CDIM + k0 + kk]);
    }
  };

  const f32x8 vz = {0, 0, 0, 0, 0, 0, 0, 0};
  f32x8 acc[6];
#pragma unroll
  for (int j = 0; j < 6; ++j) acc[j] = vz;

  stage(0, 0);
  for (int kt = 0; kt < 8; ++kt) {       // 8 k-steps of 32 over C=256
    if (kt + 1 < 8) { stage((kt + 1) & 1, (kt + 1) << 5); wait_async_5(); }
    else            { wait_async_0(); }
    __syncthreads();
    const int bu = kt & 1;
    bf16x16 a = load_a_frag(&sX[bu][wv * 16][0], 40);
#pragma unroll
    for (int j = 0; j < 6; ++j) {
      bf16x16 b = load_b_frag_t(&sWt[bu][j * 16][0], 40);
      acc[j] = wmma_bf16(a, b, acc[j]);
    }
    __syncthreads();
  }
  // spill qkv (+bias) to LDS as bf16
  {
    int rbase = wv * 16 + ((lane >> 4) << 3);
    int cl = lane & 15;
#pragma unroll
    for (int j = 0; j < 6; ++j) {
      int col = j * 16 + cl;
      float bia = qkvb[h * 96 + col];
#pragma unroll
      for (int r = 0; r < 8; ++r)
        sQKV[rbase + r][col] = (__bf16)(acc[j][r] + bia);
    }
  }
  __syncthreads();
  // S = Q K^T (scale + key mask applied on store)
  {
    bf16x16 aq = load_a_frag(&sQKV[wv * 16][0], 104);    // Q: cols 0..31
    int rbase = wv * 16 + ((lane >> 4) << 3);
#pragma unroll
    for (int j = 0; j < 4; ++j) {
      bf16x16 bk = load_b_frag_t(&sQKV[j * 16][32], 104); // K^T: Bt[key][d]
      f32x8 s = wmma_bf16(aq, bk, vz);
      int kk = j * 16 + (lane & 15);
      bool msk = ((kk + m) & 63) >= ATOK;
#pragma unroll
      for (int r = 0; r < 8; ++r)
        sS[rbase + r][kk] = msk ? -10000.0f : s[r] * SCALEF;
    }
  }
  __syncthreads();
  // per-row softmax (rows 0..63 on threads 0..63)
  if (tid < 64) {
    float mx = -3.0e38f;
    for (int k = 0; k < 64; ++k) mx = fmaxf(mx, sS[tid][k]);
    float sum = 0.0f;
    for (int k = 0; k < 64; ++k) {
      float e = __expf(sS[tid][k] - mx);
      sS[tid][k] = e;
      sum += e;
    }
    float inv = 1.0f / sum;
    for (int k = 0; k < 64; ++k) sP[tid][k] = (__bf16)(sS[tid][k] * inv);
  }
  __syncthreads();
  // O = P V, scatter rows with token t=(q+m)%64 < A
  {
    f32x8 oacc[2];
    oacc[0] = vz; oacc[1] = vz;
#pragma unroll
    for (int kh = 0; kh < 2; ++kh) {
      bf16x16 ap = load_a_frag(&sP[wv * 16][kh * 32], 72);
#pragma unroll
      for (int j = 0; j < 2; ++j) {
        bf16x16 bv = load_b_frag_rm(&sQKV[kh * 32][64 + j * 16], 104);
        oacc[j] = wmma_bf16(ap, bv, oacc[j]);
      }
    }
    int rb = wv * 16 + ((lane >> 4) << 3);
#pragma unroll
    for (int j = 0; j < 2; ++j) {
      int dcol = j * 16 + (lane & 15);
#pragma unroll
      for (int r = 0; r < 8; ++r) {
        int qrow = rb + r;
        int t = (qrow + m) & 63;
        if (t < ATOK)
          o_asy[((size_t)m * ATOK + t) * CDIM + h * DHEAD + dcol] =
              (__bf16)oacc[j][r];
      }
    }
  }
}

// ---------------- K3: generic bf16 WMMA GEMM, switchable epilogue ----------
// BwT is bf16 [N][Kdim] (pre-transposed).  64x64 block tile, 4 wave32.
// ep=1 proj:  xa = shortcut + ls1*val -> outF(f32)+outB(bf16)
// ep=2 mlp1:  gelu(val)              -> outB(bf16)
// ep=3 mlp2:  val                    -> outF(f32)
__global__ __launch_bounds__(128) void gemm_bf16_kernel(
    const __bf16* __restrict__ Amat, const __bf16* __restrict__ BwT,
    const float* __restrict__ bias, int Kdim, int N, int ep,
    const float* __restrict__ shortcut, const float* __restrict__ ls1,
    float* __restrict__ outF, __bf16* __restrict__ outB) {
  __shared__ __bf16 sA[2][64][40];
  __shared__ __bf16 sBt[2][64][40];
  const int tid = threadIdx.x, lane = tid & 31, wv = tid >> 5;
  const int rowBase = blockIdx.x * 64, colBase = blockIdx.y * 64;
  const int rh = (wv >> 1) * 32, ch = (wv & 1) * 32;
  const f32x8 vz = {0, 0, 0, 0, 0, 0, 0, 0};
  f32x8 acc[2][2] = {{vz, vz}, {vz, vz}};

  auto stage = [&](int buf, int k0) {
#pragma unroll
    for (int i = 0; i < 2; ++i) {      // A: 64 rows x 4 chunks
      int chk = i * 128 + tid, r = chk >> 2, cc = (chk & 3) * 8;
      async_ld_b128(&sA[buf][r][cc],
                    &Amat[(size_t)(rowBase + r) * Kdim + k0 + cc]);
    }
#pragma unroll
    for (int i = 0; i < 2; ++i) {      // B^T: 64 n-rows x 4 chunks
      int chk = i * 128 + tid, n = chk >> 2, kk = (chk & 3) * 8;
      async_ld_b128(&sBt[buf][n][kk],
                    &BwT[(size_t)(colBase + n) * Kdim + k0 + kk]);
    }
  };

  const int nk = Kdim >> 5;
  stage(0, 0);
  for (int kt = 0; kt < nk; ++kt) {
    if (kt + 1 < nk) { stage((kt + 1) & 1, (kt + 1) << 5); wait_async_4(); }
    else             { wait_async_0(); }
    __syncthreads();
    const int bu = kt & 1;
    bf16x16 a0 = load_a_frag(&sA[bu][rh][0], 40);
    bf16x16 a1 = load_a_frag(&sA[bu][rh + 16][0], 40);
    bf16x16 b0 = load_b_frag_t(&sBt[bu][ch][0], 40);
    bf16x16 b1 = load_b_frag_t(&sBt[bu][ch + 16][0], 40);
    acc[0][0] = wmma_bf16(a0, b0, acc[0][0]);
    acc[0][1] = wmma_bf16(a0, b1, acc[0][1]);
    acc[1][0] = wmma_bf16(a1, b0, acc[1][0]);
    acc[1][1] = wmma_bf16(a1, b1, acc[1][1]);
    __syncthreads();
  }
  const int rb = (lane >> 4) << 3, cl = lane & 15;
#pragma unroll
  for (int i = 0; i < 2; ++i)
#pragma unroll
    for (int j = 0; j < 2; ++j) {
      int col = colBase + ch + j * 16 + cl;
      float bia = bias[col];
#pragma unroll
      for (int r = 0; r < 8; ++r) {
        int row = rowBase + rh + i * 16 + rb + r;
        float val = acc[i][j][r] + bia;
        size_t oidx = (size_t)row * N + col;
        if (ep == 1) {
          float xa = shortcut[oidx] + ls1[col] * val;   // N==256 here
          outF[oidx] = xa;
          outB[oidx] = (__bf16)xa;
        } else if (ep == 2) {
          float g = 0.5f * val * (1.0f + erff(val * 0.70710678118654752f));
          outB[oidx] = (__bf16)g;
        } else {
          outF[oidx] = val;
        }
      }
    }
}

// ---------------- K4: CB group-sum reduction (deterministic 2-stage) -------
__global__ __launch_bounds__(256) void cb_partial_kernel(
    const float* __restrict__ mo, const int* __restrict__ iw,
    float* __restrict__ partial) {
  __shared__ float sacc[NGROUP][257];
  const int b = blockIdx.x, c = threadIdx.x;
#pragma unroll
  for (int g = 0; g < NGROUP; ++g) sacc[g][c] = 0.0f;
  for (int i = 0; i < 384; ++i) {           // 98304 rows / 256 blocks
    int row = b * 384 + i;
    int g = iw[row / ATOK] >> 8;            // group = window / 256
    sacc[g][c] += mo[(size_t)row * CDIM + c];
  }
#pragma unroll
  for (int g = 0; g < NGROUP; ++g)
    partial[((size_t)b * NGROUP + g) * CDIM + c] = sacc[g][c];
}

__global__ __launch_bounds__(256) void cb_reduce_kernel(
    const float* __restrict__ partial, float* __restrict__ gsum) {
  const int g = blockIdx.x, c = threadIdx.x;
  float s = 0.0f;
  for (int b = 0; b < 256; ++b)
    s += partial[((size_t)b * NGROUP + g) * CDIM + c];
  gsum[g * CDIM + c] = s;
}

// ---------------- K5: final epilogue for t<A rows --------------------------
__global__ __launch_bounds__(256) void final_epilogue_kernel(
    const float* __restrict__ xa, const float* __restrict__ mo,
    const float* __restrict__ gsum, const float* __restrict__ ls2,
    const int* __restrict__ iw, const int* __restrict__ enable_cb,
    float* __restrict__ X_out) {
  size_t idx = (size_t)blockIdx.x * 256 + threadIdx.x;   // over M*A*C
  int c = (int)(idx & 255);
  size_t ra = idx >> 8;
  int a = (int)(ra % ATOK);
  int m = (int)(ra / ATOK);
  int w = iw[m];
  float mv = mo[idx];
  if (*enable_cb) {
    int g = w >> 8;
    mv = 0.5f * mv + 0.5f * (gsum[g * CDIM + c] * (1.0f / 16384.0f));
  }
  float out = xa[idx] + ls2[c] * mv;
  X_out[(((size_t)w << 6) + a) * CDIM + c] = out;
}

// ---------------------------------------------------------------------------
extern "C" void kernel_launch(void* const* d_in, const int* in_sizes, int n_in,
                              void* d_out, int out_size, void* d_ws, size_t ws_size,
                              hipStream_t stream) {
  (void)in_sizes; (void)n_in; (void)out_size; (void)ws_size;
  const float* x      = (const float*)d_in[0];
  const int*   iw     = (const int*)d_in[1];
  const int*   en_cb  = (const int*)d_in[7];
  const float* n1g    = (const float*)d_in[8];
  const float* n1b    = (const float*)d_in[9];
  const float* n2g    = (const float*)d_in[10];
  const float* n2b    = (const float*)d_in[11];
  const float* qkv_w  = (const float*)d_in[12];
  const float* qkv_b  = (const float*)d_in[13];
  const float* proj_w = (const float*)d_in[14];
  const float* proj_b = (const float*)d_in[15];
  const float* ls1    = (const float*)d_in[16];
  const float* ls2    = (const float*)d_in[17];
  const float* mlp_w1 = (const float*)d_in[18];
  const float* mlp_b1 = (const float*)d_in[19];
  const float* mlp_w2 = (const float*)d_in[20];
  const float* mlp_b2 = (const float*)d_in[21];
  float* X_out = (float*)d_out;

  char* ws = (char*)d_ws;
  const size_t MBy = 1024ull * 1024ull;
  // persistent bf16 transposed weights [N][K]
  __bf16* wqkvT  = (__bf16*)(ws + 0);            // 768x256   384 KB
  __bf16* wprojT = (__bf16*)(ws + 384 * 1024);   // 256x256   128 KB
  __bf16* w1T    = (__bf16*)(ws + 512 * 1024);   // 1024x256  512 KB
  __bf16* w2T    = (__bf16*)(ws + 1024 * 1024);  // 256x1024  512 KB
  // lifetime-overlapped buffers
  float*  shortcut = (float*)(ws + 2 * MBy);     // 96 MB  [K1 -> proj]
  __bf16* xt       = (__bf16*)(ws + 98 * MBy);   // 64 MB  [K1 -> attn]
  __bf16* o_asy    = (__bf16*)(ws + 162 * MBy);  // 48 MB  [attn -> proj]
  float*  xa_f     = (float*)(ws + 210 * MBy);   // 96 MB  [proj -> final]
  __bf16* xa_b     = (__bf16*)(ws + 306 * MBy);  // 48 MB  [proj -> mlp1]
  __bf16* hbuf     = (__bf16*)(ws + 2 * MBy);    // 192 MB [mlp1 -> mlp2] (reuse)
  float*  mo       = (float*)(ws + 306 * MBy);   // 96 MB  [mlp2 -> final] (reuse)
  float*  partial  = (float*)(ws + 402 * MBy);   // 2 MB
  float*  gsum     = (float*)(ws + 404 * MBy);   // 8 KB

  // K0: weight convert + transpose (dst[n][k] = src[k][n])
  cvt_transpose_kernel<<<(256 * 768 + 255) / 256, 256, 0, stream>>>(qkv_w, wqkvT, 256, 768);
  cvt_transpose_kernel<<<(256 * 256 + 255) / 256, 256, 0, stream>>>(proj_w, wprojT, 256, 256);
  cvt_transpose_kernel<<<(256 * 1024 + 255) / 256, 256, 0, stream>>>(mlp_w1, w1T, 256, 1024);
  cvt_transpose_kernel<<<(1024 * 256 + 255) / 256, 256, 0, stream>>>(mlp_w2, w2T, 1024, 256);

  // K1: LN producer (also writes final rows t>=A)
  ln_gather_kernel<<<NWIN * LTOK, 256, 0, stream>>>(x, iw, n1g, n1b, n2g, n2b,
                                                    X_out, shortcut, xt);

  // K2: fused qkv + attention
  attn_kernel<<<dim3(NWIN, HEADS), 128, 0, stream>>>(xt, wqkvT, qkv_b, o_asy);

  // K3a: proj (+residual epilogue): rows = M*A = 98304
  gemm_bf16_kernel<<<dim3(98304 / 64, 256 / 64), 128, 0, stream>>>(
      o_asy, wprojT, proj_b, 256, 256, 1, shortcut, ls1, xa_f, xa_b);
  // K3b: mlp1 + gelu
  gemm_bf16_kernel<<<dim3(98304 / 64, 1024 / 64), 128, 0, stream>>>(
      xa_b, w1T, mlp_b1, 256, 1024, 2, nullptr, nullptr, nullptr, hbuf);
  // K3c: mlp2
  gemm_bf16_kernel<<<dim3(98304 / 64, 256 / 64), 128, 0, stream>>>(
      hbuf, w2T, mlp_b2, 1024, 256, 3, nullptr, nullptr, mo, nullptr);

  // K4: CB group means (deterministic)
  cb_partial_kernel<<<256, 256, 0, stream>>>(mo, iw, partial);
  cb_reduce_kernel<<<NGROUP, 256, 0, stream>>>(partial, gsum);

  // K5: final rows t<A
  final_epilogue_kernel<<<(NWIN * ATOK * CDIM) / 256, 256, 0, stream>>>(
      xa_f, mo, gsum, ls2, iw, en_cb, X_out);
}
